// SolitonAttention_54228257079950
// MI455X (gfx1250) — matrix operand order repaired and checked by reading, not compiled, not run on hardware
//
#include <hip/hip_runtime.h>

typedef __bf16 bf16_t;
typedef __attribute__((ext_vector_type(16))) __bf16 v16bf;
typedef __attribute__((ext_vector_type(8)))  __bf16 v8bf;
typedef __attribute__((ext_vector_type(4)))  __bf16 v4bf;
typedef __attribute__((ext_vector_type(8)))  float  v8f;

#define DIMD   1024
#define HEADS  16
#define BATCH  2
#define SEQ    2048
#define HD     64
#define ROWS   (BATCH * SEQ)   // 4096

// ---------------------------------------------------------------------------
// WMMA helper:  D = A(16x32 bf16) * B(32x16 bf16) + C(f32)
// ---------------------------------------------------------------------------
__device__ __forceinline__ v8f wmma_bf16(v16bf a, v16bf b, v8f c) {
  return __builtin_amdgcn_wmma_f32_16x16x32_bf16(false, a, false, b,
                                                 (short)0, c, false, false);
}

__device__ __forceinline__ v16bf concat8(v8bf lo, v8bf hi) {
  return __builtin_shufflevector(lo, hi, 0,1,2,3,4,5,6,7,8,9,10,11,12,13,14,15);
}

// A-matrix fragment (16x32 bf16), source row-major with leading dim ld.
// Lane l (l<16) holds row l, K = {hi*8..hi*8+7, 16+hi*8..23+hi*8}, hi=l>=16.
__device__ __forceinline__ v16bf load_frag_a(const bf16_t* __restrict__ base,
                                             int ld, int lane) {
  int m  = lane & 15;
  int hi = (lane >> 4) & 1;
  const bf16_t* p = base + m * ld + hi * 8;
  v8bf c0 = *(const v8bf*)(p);
  v8bf c1 = *(const v8bf*)(p + 16);
  return concat8(c0, c1);
}

// B-matrix fragment (32x16 bf16) where column n of B is row n of the source
// (i.e. B = src^T, src row-major, leading dim ld).  Lane l holds col l&15,
// K = {hi*8..hi*8+7, 16+hi*8..23+hi*8}, hi=l>=16 (same order as A).
__device__ __forceinline__ v16bf load_frag_bT(const bf16_t* __restrict__ base,
                                              int ld, int lane) {
  int n  = lane & 15;
  int hi = (lane >> 4) & 1;
  const bf16_t* p = base + n * ld + hi * 8;
  v8bf c0 = *(const v8bf*)(p);
  v8bf c1 = *(const v8bf*)(p + 16);
  return concat8(c0, c1);
}

// ---------------------------------------------------------------------------
// fp32 -> bf16 conversion (vectorized x4)
// ---------------------------------------------------------------------------
__global__ void cvt_f32_bf16_k(const float* __restrict__ in,
                               bf16_t* __restrict__ out, int n4) {
  int i = blockIdx.x * blockDim.x + threadIdx.x;
  if (i < n4) {
    float4 f = ((const float4*)in)[i];
    v4bf o;
    o[0] = (bf16_t)f.x; o[1] = (bf16_t)f.y;
    o[2] = (bf16_t)f.z; o[3] = (bf16_t)f.w;
    ((v4bf*)out)[i] = o;
  }
}

// ---------------------------------------------------------------------------
// GEMM: Y[M,N] = A[M,K](bf16) * W[N,K]^T(bf16)   (torch Linear: x @ W.T)
// Block = 256 threads = 8 waves, block tile 128x128, wave tile 32x64.
// MODE 0: row-major [M,N]           (f32 or bf16 out)
// MODE 1: [b,h,s,hd] head-split     (Q, K)
// MODE 2: [b,h,hd,s] head-split, transposed (V^T)
// ---------------------------------------------------------------------------
template <int MODE, bool F32OUT>
__global__ __launch_bounds__(256)
void gemm_xwT_k(const bf16_t* __restrict__ A, const bf16_t* __restrict__ W,
                bf16_t* __restrict__ outb, float* __restrict__ outf) {
  const int K = DIMD, N = DIMD;
  int lane = threadIdx.x & 31;
  int wave = threadIdx.x >> 5;
  int wm = wave & 3, wn = wave >> 2;
  int row0 = blockIdx.x * 128 + wm * 32;
  int col0 = blockIdx.y * 128 + wn * 64;

  v8f acc[2][4] = {};
  for (int k0 = 0; k0 < K; k0 += 32) {
    const bf16_t* pa0 = A + (size_t)row0 * K + k0;
    const bf16_t* pa1 = A + (size_t)(row0 + 16) * K + k0;
    // stream the K-tile two steps ahead into L2 ahead of the load clause (global_prefetch_b8)
    if (k0 + 64 < K) {
      __builtin_prefetch(pa0 + 64, 0, 1);
      __builtin_prefetch(pa1 + 64, 0, 1);
    }
    v16bf a0 = load_frag_a(pa0, K, lane);
    v16bf a1 = load_frag_a(pa1, K, lane);
    v16bf b[4];
#pragma unroll
    for (int j = 0; j < 4; ++j)
      b[j] = load_frag_bT(W + (size_t)(col0 + j * 16) * K + k0, K, lane);
#pragma unroll
    for (int j = 0; j < 4; ++j) {
      acc[0][j] = wmma_bf16(a0, b[j], acc[0][j]);
      acc[1][j] = wmma_bf16(a1, b[j], acc[1][j]);
    }
  }

  int nlo = lane & 15, hi = (lane >> 4) & 1;
#pragma unroll
  for (int mi = 0; mi < 2; ++mi)
#pragma unroll
    for (int j = 0; j < 4; ++j)
#pragma unroll
      for (int r = 0; r < 8; ++r) {
        int gm = row0 + mi * 16 + r + hi * 8;  // (b,s) row
        int gn = col0 + j * 16 + nlo;          // (h,d) col
        size_t off;
        if (MODE == 0) {
          off = (size_t)gm * N + gn;
        } else {
          int bb = gm >> 11, s = gm & 2047, h = gn >> 6, d = gn & 63;
          if (MODE == 1)
            off = (((size_t)(bb * HEADS + h) * SEQ) + s) * HD + d;
          else
            off = (((size_t)(bb * HEADS + h) * HD) + d) * SEQ + s;
        }
        float v = acc[mi][j][r];
        if (F32OUT) outf[off] = v;
        else        outb[off] = (bf16_t)v;
      }
}

// ---------------------------------------------------------------------------
// Fused causal soliton attention.
//   Q,K: [b,h,s,64] bf16   Vt: [b,h,64,s] bf16   O: [b,s,h*64] bf16
// grid = (SEQ/256, BATCH*HEADS), block = 256 (8 waves, 32 query rows each).
// attn p = sech^2((q.k/8 - thr)/tau) in (0,1]: streaming sum needs no max.
// Row sums computed with an extra WMMA vs an all-ones B so the normalizer
// lands in exactly the D-fragment layout of the P@V accumulators.
// Causal mask is a branchless v_cndmask select (no exec divergence).
// ---------------------------------------------------------------------------
__global__ __launch_bounds__(256)
void soliton_attn_k(const bf16_t* __restrict__ Q, const bf16_t* __restrict__ Kmat,
                    const bf16_t* __restrict__ Vt, const float* __restrict__ comp,
                    const float* __restrict__ thr, bf16_t* __restrict__ O) {
  __shared__ bf16_t plds[8][32][32];  // per-wave 32x32 P tile (bf16), 16 KB

  int lane = threadIdx.x & 31;
  int wave = threadIdx.x >> 5;
  int bh = blockIdx.y;
  int h = bh & (HEADS - 1);
  int b = bh >> 4;
  int q0 = blockIdx.x * 256 + wave * 32;

  const bf16_t* Qbh = Q    + (size_t)bh * SEQ * HD;
  const bf16_t* Kbh = Kmat + (size_t)bh * SEQ * HD;
  const bf16_t* Vbh = Vt   + (size_t)bh * HD * SEQ;

  // tau = softplus(c) + 1e-6  (v_exp2 / v_log2 hardware transcendentals)
  float c = comp[h];
  float eu = __builtin_amdgcn_exp2f(c * 1.4426950408889634f);
  float tau = __builtin_amdgcn_logf(1.0f + eu) * 0.6931471805599453f + 1e-6f;
  float inv_tau = __builtin_amdgcn_rcpf(tau);
  float th = thr[h];
  const float scale = 0.125f;  // 64^-0.5

  int nlo = lane & 15, hi = (lane >> 4) & 1;

  // Persistent Q fragments: 2 row-subtiles x 2 K-steps (hd=64)
  v16bf aQ[2][2];
#pragma unroll
  for (int mi = 0; mi < 2; ++mi)
#pragma unroll
    for (int kk = 0; kk < 2; ++kk)
      aQ[mi][kk] = load_frag_a(Qbh + (size_t)(q0 + mi * 16) * HD + kk * 32, HD, lane);

  v16bf ones;
#pragma unroll
  for (int i = 0; i < 16; ++i) ones[i] = (bf16_t)1.0f;

  v8f accO[2][4] = {};  // P@V accumulators over hd=64
  v8f rsum[2] = {};     // row-sum accumulators (P @ ones)

  for (int j0 = 0; j0 <= q0; j0 += 32) {
    // ---- S = Q K^T ---------------------------------------------------
    v16bf bK[2][2];
#pragma unroll
    for (int ni = 0; ni < 2; ++ni)
#pragma unroll
      for (int kk = 0; kk < 2; ++kk)
        bK[ni][kk] = load_frag_bT(Kbh + (size_t)(j0 + ni * 16) * HD + kk * 32, HD, lane);

#pragma unroll
    for (int mi = 0; mi < 2; ++mi)
#pragma unroll
      for (int ni = 0; ni < 2; ++ni) {
        v8f s = {};
        s = wmma_bf16(aQ[mi][0], bK[ni][0], s);
        s = wmma_bf16(aQ[mi][1], bK[ni][1], s);
        // elementwise: p = sech^2((s*scale - th)/tau); causal -> 0 via cndmask
        int kj = j0 + ni * 16 + nlo;
#pragma unroll
        for (int r = 0; r < 8; ++r) {
          int qi = q0 + mi * 16 + r + hi * 8;
          float e = (s[r] * scale - th) * inv_tau;
          // sech^2(e) = 4*e^{2e} / (e^{2e}+1)^2
          float u = __builtin_amdgcn_exp2f(e * 2.8853900817779268f);
          float d1 = u + 1.0f;
          float p = 4.0f * u * __builtin_amdgcn_rcpf(d1 * d1);
          p = (kj <= qi) ? p : 0.0f;   // branchless causal mask
          plds[wave][mi * 16 + r + hi * 8][ni * 16 + nlo] = (bf16_t)p;
        }
      }

    // per-wave LDS write -> read ordering
    asm volatile("s_wait_dscnt 0" ::: "memory");

    // ---- P @ V (B fragments from pre-transposed V) -------------------
    v16bf bV[4];
#pragma unroll
    for (int ni = 0; ni < 4; ++ni)
      bV[ni] = load_frag_bT(Vbh + (size_t)(ni * 16) * SEQ + j0, SEQ, lane);

#pragma unroll
    for (int mi = 0; mi < 2; ++mi) {
      v16bf pa = load_frag_a(&plds[wave][mi * 16][0], 32, lane);
      rsum[mi] = wmma_bf16(pa, ones, rsum[mi]);
#pragma unroll
      for (int ni = 0; ni < 4; ++ni)
        accO[mi][ni] = wmma_bf16(pa, bV[ni], accO[mi][ni]);
    }
  }

  // ---- normalize and write O[b, s, h*64 + d] -------------------------
#pragma unroll
  for (int mi = 0; mi < 2; ++mi)
#pragma unroll
    for (int r = 0; r < 8; ++r) {
      float inv = __builtin_amdgcn_rcpf(rsum[mi][r] + 1e-6f);
      int s = q0 + mi * 16 + r + hi * 8;
      size_t rowoff = ((size_t)b * SEQ + s) * DIMD + h * HD;
#pragma unroll
      for (int ni = 0; ni < 4; ++ni)
        O[rowoff + ni * 16 + nlo] = (bf16_t)(accO[mi][ni][r] * inv);
    }
}

// ---------------------------------------------------------------------------
// launcher
// ---------------------------------------------------------------------------
extern "C" void kernel_launch(void* const* d_in, const int* in_sizes, int n_in,
                              void* d_out, int out_size, void* d_ws, size_t ws_size,
                              hipStream_t stream) {
  const float* x    = (const float*)d_in[0];
  const float* Wq   = (const float*)d_in[1];
  const float* Wk   = (const float*)d_in[2];
  const float* Wv   = (const float*)d_in[3];
  const float* Wo   = (const float*)d_in[4];
  const float* comp = (const float*)d_in[5];
  const float* thr  = (const float*)d_in[6];
  float* out = (float*)d_out;

  // workspace carve-up (48 MB total)
  char* ws = (char*)d_ws;
  bf16_t* xb  = (bf16_t*)ws;  ws += (size_t)ROWS * DIMD * 2;   // 8 MB
  bf16_t* wqb = (bf16_t*)ws;  ws += (size_t)DIMD * DIMD * 2;   // 2 MB
  bf16_t* wkb = (bf16_t*)ws;  ws += (size_t)DIMD * DIMD * 2;
  bf16_t* wvb = (bf16_t*)ws;  ws += (size_t)DIMD * DIMD * 2;
  bf16_t* wob = (bf16_t*)ws;  ws += (size_t)DIMD * DIMD * 2;
  bf16_t* qb  = (bf16_t*)ws;  ws += (size_t)ROWS * DIMD * 2;   // [b,h,s,hd]
  bf16_t* kb  = (bf16_t*)ws;  ws += (size_t)ROWS * DIMD * 2;   // [b,h,s,hd]
  bf16_t* vtb = (bf16_t*)ws;  ws += (size_t)ROWS * DIMD * 2;   // [b,h,hd,s]
  bf16_t* ob  = (bf16_t*)ws;  ws += (size_t)ROWS * DIMD * 2;   // [b,s,h*hd]

  // 1) fp32 -> bf16 conversions
  {
    int n4 = ROWS * DIMD / 4;
    cvt_f32_bf16_k<<<(n4 + 255) / 256, 256, 0, stream>>>(x, xb, n4);
    int w4 = DIMD * DIMD / 4;
    cvt_f32_bf16_k<<<(w4 + 255) / 256, 256, 0, stream>>>(Wq, wqb, w4);
    cvt_f32_bf16_k<<<(w4 + 255) / 256, 256, 0, stream>>>(Wk, wkb, w4);
    cvt_f32_bf16_k<<<(w4 + 255) / 256, 256, 0, stream>>>(Wv, wvb, w4);
    cvt_f32_bf16_k<<<(w4 + 255) / 256, 256, 0, stream>>>(Wo, wob, w4);
  }

  // 2) projections (WMMA GEMMs)
  dim3 gg(ROWS / 128, DIMD / 128);
  gemm_xwT_k<1, false><<<gg, 256, 0, stream>>>(xb, wqb, qb, nullptr);
  gemm_xwT_k<1, false><<<gg, 256, 0, stream>>>(xb, wkb, kb, nullptr);
  gemm_xwT_k<2, false><<<gg, 256, 0, stream>>>(xb, wvb, vtb, nullptr);

  // 3) fused causal soliton attention
  dim3 ga(SEQ / 256, BATCH * HEADS);
  soliton_attn_k<<<ga, 256, 0, stream>>>(qb, kb, vtb, comp, thr, ob);

  // 4) output projection -> fp32
  gemm_xwT_k<0, true><<<gg, 256, 0, stream>>>(ob, wob, nullptr, out);
}
